// QuadtreePartitionBasedContextModel_90555090469403
// MI455X (gfx1250) — matrix operand order repaired and checked by reading, not compiled, not run on hardware
//
#include <hip/hip_runtime.h>
#include <hip/hip_bf16.h>
#include <math.h>

typedef __bf16 bf16;
typedef __attribute__((ext_vector_type(16))) __bf16 v16bf;
typedef __attribute__((ext_vector_type(8)))  float   v8f;

union Frag {
  struct { uint4 lo, hi; } q;
  v16bf v;
};

#define NTOT   65536     // B*H*W = 4*128*128 pixels
#define HW     16384
#define GT_N   256       // pixels per workgroup tile
#define GT_CO  64        // out-channels per workgroup tile
#define ROWB   80        // LDS row stride (64B data + pad, 16B aligned)
#define XBYTES (GT_N * ROWB)             // 20480
#define LDS_HALF (XBYTES + GT_CO * ROWB) // 25600 (X tile + W tile)
#define NBUF   3                         // triple-buffered pipeline

__device__ __constant__ int d_MASK_PERM[4][4] = {{0,1,2,3},{3,2,1,0},{2,3,0,1},{1,0,3,2}};
__device__ __constant__ int d_MEAN_PERM[4][4] = {{0,1,2,3},{0,1,2,3},{2,3,0,1},{1,0,3,2}};

__device__ __forceinline__ void async_b128(unsigned lds_off, const void* gaddr) {
  // CDNA5 async copy: cache/memory -> LDS, tracked by ASYNCcnt (no VGPR round trip)
  asm volatile("global_load_async_to_lds_b128 %0, %1, off"
               :: "v"(lds_off), "v"(gaddr) : "memory");
}
__device__ __forceinline__ void wait_async0() {
  asm volatile("s_wait_asynccnt 0" ::: "memory");
}
__device__ __forceinline__ void wait_async5() {
  // allow the newest stage (5 copies) to remain in flight; oldest chunk is resident
  asm volatile("s_wait_asynccnt 5" ::: "memory");
}

// ---------------------------------------------------------------------------
// Fused 1x1-conv GEMM:  out[n][co] = act( sum_ci X[n][ci]*W[co][ci] + bias[co] )
//                                      (+ Res[n][co])
// A (WMMA SRC0) = W tile (M=co, K=ci), B (SRC1) = X tile (K=ci, N=pixel).
// Triple-buffered LDS, async global->LDS staging two chunks ahead,
// 8 WMMAs / wave / K-step, one barrier per K-step.
// ---------------------------------------------------------------------------
__global__ __launch_bounds__(256)
void gemm_wmma(const bf16* __restrict__ X, const bf16* __restrict__ W,
               const float* __restrict__ bias, const bf16* __restrict__ Res,
               float slope, bf16* __restrict__ outB, float* __restrict__ outF,
               int Cin, int Cout)
{
  __shared__ __align__(16) unsigned char lds[NBUF * LDS_HALF];   // 76800 B

  const int tid  = threadIdx.x;
  const int lane = tid & 31;
  const int wave = tid >> 5;          // 8 waves: 2 (co) x 4 (n)
  const int wco  = wave & 1;
  const int wn   = wave >> 1;
  const int lr   = lane & 15;
  const int kh   = lane >> 4;         // K half selector
  const size_t nBase  = (size_t)blockIdx.x * GT_N;
  const int    coBase = blockIdx.y * GT_CO;
  const unsigned ldsBase = (unsigned)(size_t)&lds[0];

  v8f acc[2][4];
  #pragma unroll
  for (int i = 0; i < 2; ++i)
    #pragma unroll
    for (int j = 0; j < 4; ++j)
      #pragma unroll
      for (int e = 0; e < 8; ++e) acc[i][j][e] = 0.0f;

  // Issue async stage of K-chunk (k0) into LDS buffer `buf`:
  // X: 256 rows x 64B (1 row per thread, 4x b128); W: 64 rows x 64B (1 b128/thread)
  auto stage = [&](int k0, int buf) {
    unsigned base = ldsBase + buf * LDS_HALF;
    const bf16* gx = X + (nBase + (size_t)tid) * (size_t)Cin + k0;
    unsigned lx = base + tid * ROWB;
    #pragma unroll
    for (int seg = 0; seg < 4; ++seg)
      async_b128(lx + seg * 16, gx + seg * 8);
    int wrow = tid >> 2, wseg = tid & 3;
    const bf16* gw = W + (size_t)(coBase + wrow) * (size_t)Cin + k0 + wseg * 8;
    async_b128(base + XBYTES + wrow * ROWB + wseg * 16, gw);
  };

  const int nk = Cin >> 5;
  stage(0, 0);                          // prologue: two chunks in flight
  if (nk > 1) stage(32, 1);

  int cur = 0;                          // buffer holding chunk k
  for (int k = 0; k < nk; ++k) {
    if (k + 1 < nk) wait_async5();      // chunk k resident; k+1 may still fly
    else            wait_async0();      // last chunk: everything must be resident
    __syncthreads();                    // ... and visible to all waves
    if (k + 2 < nk) {                   // keep two chunks of lookahead
      int nb = cur + 2; if (nb >= NBUF) nb -= NBUF;
      stage((k + 2) * 32, nb);
    }

    const unsigned char* bufp = lds + cur * LDS_HALF;
    Frag A[2], Bf[4];
    #pragma unroll
    for (int ct = 0; ct < 2; ++ct) {    // A = weights, documented 16-bit A layout
      const unsigned char* p = bufp + XBYTES + (wco * 32 + ct * 16 + lr) * ROWB + kh * 16;
      A[ct].q.lo = *(const uint4*)p;            // K = 8*kh .. 8*kh+7
      A[ct].q.hi = *(const uint4*)(p + 32);     // K = 16+8*kh .. 23+8*kh
    }
    #pragma unroll
    for (int nt = 0; nt < 4; ++nt) {    // B = activations, documented 16-bit layout
      const unsigned char* p = bufp + (wn * 64 + nt * 16 + lr) * ROWB + kh * 16;
      Bf[nt].q.lo = *(const uint4*)p;           // K = 8*kh .. 8*kh+7
      Bf[nt].q.hi = *(const uint4*)(p + 32);    // K = 16+8*kh .. 23+8*kh
    }
    #pragma unroll
    for (int ct = 0; ct < 2; ++ct)
      #pragma unroll
      for (int nt = 0; nt < 4; ++nt)
        acc[ct][nt] = __builtin_amdgcn_wmma_f32_16x16x32_bf16(
            false, A[ct].v, false, Bf[nt].v, (short)0, acc[ct][nt], false, false);

    cur = (cur + 1 == NBUF) ? 0 : cur + 1;
  }

  // ---- epilogue: bias + leaky + residual + store ----
  #pragma unroll
  for (int ct = 0; ct < 2; ++ct) {
    int co0 = coBase + wco * 32 + ct * 16 + kh * 8;
    float4 b0 = *(const float4*)(bias + co0);
    float4 b1 = *(const float4*)(bias + co0 + 4);
    #pragma unroll
    for (int nt = 0; nt < 4; ++nt) {
      size_t n = nBase + wn * 64 + nt * 16 + lr;
      float r[8];
      v8f a = acc[ct][nt];
      r[0]=a[0]+b0.x; r[1]=a[1]+b0.y; r[2]=a[2]+b0.z; r[3]=a[3]+b0.w;
      r[4]=a[4]+b1.x; r[5]=a[5]+b1.y; r[6]=a[6]+b1.z; r[7]=a[7]+b1.w;
      #pragma unroll
      for (int e = 0; e < 8; ++e) r[e] = r[e] > 0.0f ? r[e] : r[e] * slope;
      if (Res) {
        uint4 rv = *(const uint4*)(Res + n * (size_t)Cout + co0);
        const bf16* rp = (const bf16*)&rv;
        #pragma unroll
        for (int e = 0; e < 8; ++e) r[e] += (float)rp[e];
      }
      if (outF) {
        float* op = outF + n * (size_t)Cout + co0;
        *(float4*)(op)     = make_float4(r[0], r[1], r[2], r[3]);
        *(float4*)(op + 4) = make_float4(r[4], r[5], r[6], r[7]);
      } else {
        uint4 ov; bf16* op = (bf16*)&ov;
        #pragma unroll
        for (int e = 0; e < 8; ++e) op[e] = (bf16)r[e];
        *(uint4*)(outB + n * (size_t)Cout + co0) = ov;
      }
    }
  }
}

// Depthwise 3x3 SAME conv, bf16 [n][384] -> bf16 [n][384], f32 weights (c,1,3,3)
__global__ __launch_bounds__(256)
void dwconv3(const bf16* __restrict__ in, const float* __restrict__ wd,
             const float* __restrict__ bd, bf16* __restrict__ out)
{
  size_t i = (size_t)blockIdx.x * 256 + threadIdx.x;     // over NTOT*384
  if (i >= (size_t)NTOT * 384) return;
  int c = (int)(i % 384);
  size_t n = i / 384;
  int b = (int)(n >> 14), pix = (int)(n & 16383);
  int h = pix >> 7, w = pix & 127;
  float acc = bd[c];
  #pragma unroll
  for (int dh = -1; dh <= 1; ++dh) {
    int hh = h + dh;
    if (hh < 0 || hh > 127) continue;
    #pragma unroll
    for (int dw = -1; dw <= 1; ++dw) {
      int ww = w + dw;
      if (ww < 0 || ww > 127) continue;
      size_t src = (((size_t)b << 14) + (hh << 7) + ww) * 384 + c;
      acc += (float)in[src] * wd[c * 9 + (dh + 1) * 3 + (dw + 1)];
    }
  }
  out[n * 384 + c] = (bf16)acc;
}

__global__ __launch_bounds__(256)
void cast_f32_bf16(const float* __restrict__ in, bf16* __restrict__ out, size_t n)
{
  size_t i = (size_t)blockIdx.x * 256 + threadIdx.x;
  if (i < n) out[i] = (bf16)in[i];
}

// condition NCHW f32 [4,256,128,128] -> actA channels 128..383 (bf16, [n][384])
__global__ __launch_bounds__(256)
void cond_to_act(const float* __restrict__ cond, bf16* __restrict__ actA)
{
  size_t i = (size_t)blockIdx.x * 256 + threadIdx.x;     // 4*256*16384 = 2^24
  if (i >= ((size_t)1 << 24)) return;
  size_t p = i & 16383, c = (i >> 14) & 255, b = i >> 22;
  actA[(b * HW + p) * 384 + 128 + c] = (bf16)cond[i];
}

// Per-step masked round + Gaussian likelihood. Updates y_hat (f32 NCHW + bf16 [n][384])
// and accumulates likelihood into lkd (NCHW).
__global__ __launch_bounds__(256)
void step_elem(const float* __restrict__ y, const float* __restrict__ cond,
               const float* __restrict__ prior, float* __restrict__ yhat,
               float* __restrict__ lkd, bf16* __restrict__ actA, int step)
{
  size_t i = (size_t)blockIdx.x * 256 + threadIdx.x;     // 4*128*16384 = 2^23
  if (i >= ((size_t)1 << 23)) return;
  int pix = (int)(i & 16383);
  int kc  = (int)((i >> 14) & 127);
  int b   = (int)(i >> 21);
  int h = pix >> 7, w = pix & 127;
  int par = ((h & 1) << 1) | (w & 1);
  int k = kc >> 5, c = kc & 31;
  if (d_MASK_PERM[step][k] != par) return;
  int mc = d_MEAN_PERM[step][k] * 32 + c;

  float yv = y[i];
  float s, m;
  if (step == 0) {
    const float* cb = cond + (size_t)b * 256 * HW;
    s = cb[(size_t)(128 + kc) * HW + pix];                 // scales = 2nd half
    m = cb[(size_t)mc * HW + pix];
  } else {
    const float* pr = prior + ((size_t)b * HW + pix) * 256;
    s = pr[kc];                                            // ch[0..3]  = scales
    m = pr[128 + mc];                                      // ch[4..7]  = means
  }
  float yh = rintf(yv - m) + m;
  float nv = yhat[i] + yh;
  yhat[i] = nv;
  actA[((size_t)b * HW + pix) * 384 + kc] = (bf16)nv;

  float sc = fmaxf(fabsf(s), 0.11f);
  float v  = rintf(yv) - m;
  float l  = normcdff((v + 0.5f) / sc) - normcdff((v - 0.5f) / sc);
  lkd[i] += l;
}

// ---------------------------------------------------------------------------
extern "C" void kernel_launch(void* const* d_in, const int* in_sizes, int n_in,
                              void* d_out, int out_size, void* d_ws, size_t ws_size,
                              hipStream_t stream)
{
  (void)in_sizes; (void)n_in; (void)out_size; (void)ws_size;
  const float* d_input = (const float*)d_in[0];
  const float* d_cond  = (const float*)d_in[1];

  // pytree leaf order (sorted dict keys): adaptors, then prior blocks
  const int aB[3] = {2, 4, 6};
  const int aW[3] = {3, 5, 7};
  const int b_b1[3] = {8, 18, 28}, b_b2[3] = {9, 19, 29};
  const int b_bd[3] = {10, 20, 31}, b_w1[3] = {11, 21, 32};
  const int b_w2[3] = {12, 22, 33}, b_wd[3] = {13, 23, 35};
  const int b_fb1[3] = {14, 24, 36}, b_fb2[3] = {15, 25, 37};
  const int b_fw1[3] = {16, 26, 38}, b_fw2[3] = {17, 27, 39};
  const int b_ba = 30, b_wa = 34;
  const int coBlk[3] = {384, 384, 256};

  char* ws = (char*)d_ws;
  size_t off = 0;
  auto alloc = [&](size_t bytes) { size_t o = off; off += (bytes + 255) & ~(size_t)255; return o; };

  bf16* actA = (bf16*)(ws + alloc((size_t)NTOT * 384 * 2));  // [yhat | condition]
  bf16* buf0 = (bf16*)(ws + alloc((size_t)NTOT * 384 * 2));
  bf16* buf1 = (bf16*)(ws + alloc((size_t)NTOT * 384 * 2));
  bf16* buf2 = (bf16*)(ws + alloc((size_t)NTOT * 384 * 2));
  bf16* bufH = (bf16*)(ws + alloc((size_t)NTOT * 1024 * 2));
  float* bufP = (float*)(ws + alloc((size_t)NTOT * 256 * 4));
  bf16* bufI = (bf16*)(ws + alloc((size_t)NTOT * 256 * 2));

  auto convW = [&](int idx, size_t count) -> bf16* {
    bf16* dst = (bf16*)(ws + alloc(count * 2));
    cast_f32_bf16<<<(unsigned)((count + 255) / 256), 256, 0, stream>>>((const float*)d_in[idx], dst, count);
    return dst;
  };

  bf16 *wA[3], *wW1[3], *wW2[3], *wF1[3], *wF2[3], *wWa = nullptr;
  for (int s = 0; s < 3; ++s) wA[s] = convW(aW[s], (size_t)384 * 384);
  for (int blk = 0; blk < 3; ++blk) {
    int co = coBlk[blk];
    wW1[blk] = convW(b_w1[blk], (size_t)384 * 384);
    wW2[blk] = convW(b_w2[blk], (size_t)co * 384);
    wF1[blk] = convW(b_fw1[blk], (size_t)1024 * co);
    wF2[blk] = convW(b_fw2[blk], (size_t)co * 1024);
  }
  wWa = convW(b_wa, (size_t)256 * 384);

  float* yhatO = (float*)d_out;
  float* lkdO  = yhatO + ((size_t)1 << 23);

  hipMemsetAsync(d_out, 0, ((size_t)1 << 24) * 4, stream);
  hipMemsetAsync(actA, 0, (size_t)NTOT * 384 * 2, stream);
  cond_to_act<<<65536, 256, 0, stream>>>(d_cond, actA);

  auto gemm = [&](const bf16* X, const bf16* Wt, const float* bias, const bf16* Res,
                  float slope, bf16* oB, float* oF, int Cin, int Cout) {
    dim3 g(NTOT / GT_N, Cout / GT_CO);
    gemm_wmma<<<g, 256, 0, stream>>>(X, Wt, bias, Res, slope, oB, oF, Cin, Cout);
  };

  // step 0: scales/means come straight from condition
  step_elem<<<32768, 256, 0, stream>>>(d_input, d_cond, bufP, yhatO, lkdO, actA, 0);

  for (int step = 1; step <= 3; ++step) {
    // adaptor: actA[384] -> buf0[384]
    gemm(actA, wA[step - 1], (const float*)d_in[aB[step - 1]], nullptr, 1.0f, buf0, nullptr, 384, 384);

    for (int blk = 0; blk < 3; ++blk) {
      int co = coBlk[blk];
      // dc.w1 + leaky(0.01)
      gemm(buf0, wW1[blk], (const float*)d_in[b_b1[blk]], nullptr, 0.01f, buf1, nullptr, 384, 384);
      // depthwise 3x3
      dwconv3<<<(unsigned)(((size_t)NTOT * 384 + 255) / 256), 256, 0, stream>>>(
          buf1, (const float*)d_in[b_wd[blk]], (const float*)d_in[b_bd[blk]], buf2);
      // identity path
      const bf16* res = buf0;
      if (blk == 2) {
        gemm(buf0, wWa, (const float*)d_in[b_ba], nullptr, 1.0f, bufI, nullptr, 384, 256);
        res = bufI;
      }
      // dc.w2 + identity add -> o in buf1 (co channels)
      gemm(buf2, wW2[blk], (const float*)d_in[b_b2[blk]], res, 1.0f, buf1, nullptr, 384, co);
      // ffn.w1 + leaky(0.1) -> bufH[1024]
      gemm(buf1, wF1[blk], (const float*)d_in[b_fb1[blk]], nullptr, 0.1f, bufH, nullptr, co, 1024);
      // ffn.w2 + leaky(0.1) + o  -> next x (bf16) or final prior output (f32)
      if (blk < 2)
        gemm(bufH, wF2[blk], (const float*)d_in[b_fb2[blk]], buf1, 0.1f, buf0, nullptr, 1024, co);
      else
        gemm(bufH, wF2[blk], (const float*)d_in[b_fb2[blk]], buf1, 0.1f, nullptr, bufP, 1024, co);
    }
    step_elem<<<32768, 256, 0, stream>>>(d_input, d_cond, bufP, yhatO, lkdO, actA, step);
  }
}
